// NearestNeigborClassifier_35201551958756
// MI455X (gfx1250) — matrix operand order repaired and checked by reading, not compiled, not run on hardware
//
#include <hip/hip_runtime.h>

typedef __attribute__((ext_vector_type(16))) __bf16 v16bf;
typedef __attribute__((ext_vector_type(16))) float  f32x16;
typedef __attribute__((ext_vector_type(8)))  float  f32x8, v8f;
typedef __attribute__((ext_vector_type(4)))  float  f32x4;

#define ND 2048
#define NC 1000
#define ROWS_PER_BLOCK 128   // 8 waves x 16 rows

__global__ __launch_bounds__(256)
void nnc_fused_kernel(const float* __restrict__ x,
                      const float* __restrict__ means,
                      float* __restrict__ out)
{
  // Double-buffered staged B super-tile:
  // [2][8 class-tiles][16 classes][32 K] bf16 = 2 x 8 KB
  __shared__ __align__(32) __bf16 Bs[2][4096];

  const int tid  = threadIdx.x;
  const int lane = tid & 31;
  const int wave = tid >> 5;
  const int nlo  = lane & 15;   // column / class-in-tile
  const int ksel = lane >> 4;   // K half selector
  const int rowBase = blockIdx.x * ROWS_PER_BLOCK + wave * 16;

  const float* xrow = x + (size_t)(rowBase + nlo) * ND;

  // cooperative staging role: class-in-supertile and K half
  const int scc  = tid >> 1;                                        // 0..127
  const int skh  = tid & 1;                                         // 0..1
  const int sdst = (scc >> 4) * 512 + (scc & 15) * 32 + skh * 16;   // bf16 elems
  const int boff = nlo * 32 + ksel * 16;                            // B read offset

  float best[8];
  int   bidx[8];
  #pragma unroll
  for (int j = 0; j < 8; ++j) { best[j] = -3.402823466e38f; bidx[j] = 0; }

  for (int s = 0; s < 8; ++s) {                // 8 super-tiles of 128 classes
    const int sBase = s * 128;
    int sc = sBase + scc; if (sc > NC - 1) sc = NC - 1;   // clamp tail reads
    const float* mrow = means + (size_t)sc * ND + skh * 16;

    v8f acc[8];
    #pragma unroll
    for (int t = 0; t < 8; ++t)
      #pragma unroll
      for (int j = 0; j < 8; ++j) acc[t][j] = 0.0f;

    for (int kc = 0; kc < ND; kc += 32) {
      const int p = (kc >> 5) & 1;   // LDS buffer parity

      // ---- stage B chunk (fp32 -> bf16, packed cvts) into LDS ----
      {
        f32x16 mf = *(const f32x16*)(mrow + kc);          // 64B, 4x b128
        v16bf  mb = __builtin_convertvector(mf, v16bf);   // v_cvt_pk_bf16_f32 x8
        *(v16bf*)&Bs[p][sdst] = mb;                       // 32B ds store
      }

      // ---- A operand: 16 rows x 32 K, contiguous K (k = 16*half + s), from global ----
      f32x8 alo = *(const f32x8*)(xrow + kc + ksel * 16);
      f32x8 ahi = *(const f32x8*)(xrow + kc + ksel * 16 + 8);
      f32x16 af = __builtin_shufflevector(alo, ahi,
                    0,1,2,3,4,5,6,7,8,9,10,11,12,13,14,15);
      v16bf A = __builtin_convertvector(af, v16bf);

      __syncthreads();   // staged chunk visible; prior buffer reads retired

      // ---- preload all 8 B operands, then 8 back-to-back WMMAs ----
      v16bf Bt[8];
      #pragma unroll
      for (int t = 0; t < 8; ++t)
        Bt[t] = *(const v16bf*)&Bs[p][t * 512 + boff];    // 2x ds_load_b128 each

      #pragma unroll
      for (int t = 0; t < 8; ++t)
        acc[t] = __builtin_amdgcn_wmma_f32_16x16x32_bf16(
                     /*neg_a=*/false, A, /*neg_b=*/false, Bt[t],
                     /*c_mod=*/(short)0, acc[t],
                     /*reuse_a=*/false, /*reuse_b=*/false);
    }

    // fold this super-tile's scores into the running argmax
    #pragma unroll
    for (int t = 0; t < 8; ++t) {
      const int c = sBase + t * 16 + nlo;
      const bool valid = (c < NC);
      #pragma unroll
      for (int j = 0; j < 8; ++j) {
        float scv = valid ? acc[t][j] : -3.402823466e38f;
        if (scv > best[j]) { best[j] = scv; bidx[j] = c; }  // strict > keeps lowest idx
      }
    }
  }

  // cross-lane argmax reduction within each 16-lane half
  #pragma unroll
  for (int off = 8; off >= 1; off >>= 1) {
    #pragma unroll
    for (int j = 0; j < 8; ++j) {
      float ob = __shfl_xor(best[j], off, 16);
      int   oi = __shfl_xor(bidx[j], off, 16);
      if (ob > best[j] || (ob == best[j] && oi < bidx[j])) {
        best[j] = ob; bidx[j] = oi;
      }
    }
  }

  // one-hot output: each wave writes its 16 rows fully (zeros + single 1.0)
  #pragma unroll
  for (int r = 0; r < 16; ++r) {
    const int tgt = __shfl(bidx[r & 7], (r >> 3) * 16, 32);
    float* orow = out + (size_t)(rowBase + r) * NC;
    for (int i = lane; i < NC / 4; i += 32) {   // 250 float4 per row, 16B aligned
      const int b4 = i * 4;
      f32x4 v;
      v.x = (tgt == b4    ) ? 1.0f : 0.0f;
      v.y = (tgt == b4 + 1) ? 1.0f : 0.0f;
      v.z = (tgt == b4 + 2) ? 1.0f : 0.0f;
      v.w = (tgt == b4 + 3) ? 1.0f : 0.0f;
      *(f32x4*)(orow + b4) = v;
    }
  }
}

extern "C" void kernel_launch(void* const* d_in, const int* in_sizes, int n_in,
                              void* d_out, int out_size, void* d_ws, size_t ws_size,
                              hipStream_t stream) {
  const float* x     = (const float*)d_in[0];   // [16384, 2048] fp32
  const float* means = (const float*)d_in[1];   // [1000, 2048] fp32
  float* out = (float*)d_out;                   // [16384, 1000] fp32 one-hot
  (void)in_sizes; (void)n_in; (void)d_ws; (void)ws_size; (void)out_size;

  dim3 grid(16384 / ROWS_PER_BLOCK);  // 128 workgroups
  dim3 block(256);                    // 8 wave32s
  nnc_fused_kernel<<<grid, block, 0, stream>>>(x, means, out);
}
